// simcomen_17712445129475
// MI455X (gfx1250) — matrix operand order repaired and checked by reading, not compiled, not run on hardware
//
#include <hip/hip_runtime.h>
#include <cstdint>
#include <cstddef>

#define GENES 512

typedef __attribute__((ext_vector_type(16))) _Float16 v16h;
typedef __attribute__((ext_vector_type(8)))  float    v8f;

union HalfVec { v16h v; uint4 q[2]; };

// ---------------- f32 -> f16 conversion (weights) ----------------
__global__ void k_cvt_f16(const float* __restrict__ src, _Float16* __restrict__ dst, int n) {
    int i = blockIdx.x * blockDim.x + threadIdx.x;
    if (i < n) dst[i] = (_Float16)src[i];
}

// ---------------- gex: wave-per-row cumulative sin product ----------------
// gex[r,i] = (prod_{j<i} sin sphex[r,j]) * (i < G-1 ? cos sphex[r,i] : 1), nan_to_num
__global__ void k_gex(const float* __restrict__ sphex, _Float16* __restrict__ gexh, int nspots) {
    const int lane = threadIdx.x & 31;
    const int wave = threadIdx.x >> 5;
    const int row  = blockIdx.x * (blockDim.x >> 5) + wave;
    if (row >= nspots) return;
    const float* xr = sphex + (size_t)row * (GENES - 1);

    float lp[16], cc[16];
    float p = 1.0f;
#pragma unroll
    for (int t = 0; t < 16; ++t) {
        int i = lane * 16 + t;
        float s, c;
        if (i < GENES - 1) {
            __sincosf(xr[i], &s, &c);
        } else { s = 1.0f; c = 1.0f; }
        lp[t] = p;          // exclusive prefix within the lane
        cc[t] = c;
        p *= s;
    }
    // multiplicative inclusive scan of per-lane products across wave32
    float inc = p;
#pragma unroll
    for (int d = 1; d < 32; d <<= 1) {
        float o = __shfl_up(inc, d, 32);
        if (lane >= d) inc *= o;
    }
    float exc = __shfl_up(inc, 1, 32);
    if (lane == 0) exc = 1.0f;

    _Float16* out = gexh + (size_t)row * GENES + lane * 16;
#pragma unroll
    for (int t = 0; t < 16; ++t) {
        float g = exc * lp[t] * cc[t];
        if (!__builtin_isfinite(g)) g = 0.0f;
        out[t] = (_Float16)g;
    }
}

// ---------------- WMMA GEMM: out[m,n] = sum_k A[m,k] * W[n,k] (+ bias[n]) ----------------
// A: [M, 512] f16 row-major; W: [512, 512] f16 row-major (so B[k,n] = W[n,k]).
// block = 128 threads = 4 waves; each wave owns a 16x64 strip (4 accumulators
// sharing one A fragment per k-step); grid = (M/16, 512/256).
template <typename OutT>
__global__ void k_gemm_wmma(const _Float16* __restrict__ A, const _Float16* __restrict__ Bw,
                            const float* __restrict__ bias, OutT* __restrict__ out, int M) {
    const int lane = threadIdx.x & 31;
    const int wave = threadIdx.x >> 5;
    const int m0 = blockIdx.x * 16;
    const int n0 = blockIdx.y * 256 + wave * 64;     // 64-column strip per wave
    if (m0 >= M) return;

    const int hi  = lane >> 4;     // 0: low half-wave, 1: high half-wave
    const int l15 = lane & 15;
    int m = m0 + l15; if (m >= M) m = M - 1;         // clamp (M % 16 == 0 in practice)

    const _Float16* arow = A  + (size_t)m * GENES;
    const _Float16* brow = Bw + (size_t)(n0 + l15) * GENES;   // sub-tiles at row offsets 0,16,32,48

    v8f acc[4] = {{}, {}, {}, {}};
#pragma unroll 2
    for (int k0 = 0; k0 < GENES; k0 += 32) {
        __builtin_prefetch(arow + k0 + 64, 0, 1);
        // A 16x32 f16 layout: lanes 0-15 -> K {0..7,16..23}, lanes 16-31 -> K {8..15,24..31}
        HalfVec a;
        a.q[0] = *(const uint4*)(arow + k0 + hi * 8);
        a.q[1] = *(const uint4*)(arow + k0 + 16 + hi * 8);
        // B 32x16 f16 layout: lanes 0-15 -> K {0..7,16..23}, lanes 16-31 -> K {8..15,24..31}, N = lane&15
        const int koff = k0 + hi * 8;
        HalfVec b[4];
#pragma unroll
        for (int t = 0; t < 4; ++t) {
            const _Float16* bp = brow + (size_t)(t * 16) * GENES + koff;
            b[t].q[0] = *(const uint4*)(bp);
            b[t].q[1] = *(const uint4*)(bp + 16);
        }
#pragma unroll
        for (int t = 0; t < 4; ++t) {
            acc[t] = __builtin_amdgcn_wmma_f32_16x16x32_f16(
                /*neg_a=*/false, a.v, /*neg_b=*/false, b[t].v,
                /*c_mod=*/(short)0, acc[t], /*reuse_a=*/false, /*reuse_b=*/false);
        }
    }

    const bool full = (m0 + 16 <= M);    // wave-uniform: tile fully in range
#pragma unroll
    for (int t = 0; t < 4; ++t) {
        const int n = n0 + t * 16 + l15;
        const float bn = bias ? bias[n] : 0.0f;
        OutT* op = out + (size_t)(m0 + hi * 8) * GENES + n;
        if (full) {
#pragma unroll
            for (int v = 0; v < 8; ++v)                 // C/D layout: VGPR v -> M = v + 8*hi
                op[(size_t)v * GENES] = (OutT)(acc[t][v] + bn);
        } else {
#pragma unroll
            for (int v = 0; v < 8; ++v)
                if (m0 + hi * 8 + v < M)
                    op[(size_t)v * GENES] = (OutT)(acc[t][v] + bn);
        }
    }
}

// ---------------- graph helpers ----------------
__global__ void k_zero(float* __restrict__ p, int n) {
    int i = blockIdx.x * blockDim.x + threadIdx.x;
    if (i < n) p[i] = 0.0f;
}

__global__ void k_deg(const int* __restrict__ dst, float* __restrict__ deg, int E) {
    int e = blockIdx.x * blockDim.x + threadIdx.x;
    if (e < E) atomicAdd(&deg[dst[e]], 1.0f);
}

__global__ void k_dinv(const float* __restrict__ deg, float* __restrict__ dinv, int n) {
    int i = blockIdx.x * blockDim.x + threadIdx.x;
    if (i < n) {
        float d = deg[i];
        dinv[i] = d > 0.0f ? rsqrtf(d) : 0.0f;
    }
}

__global__ void k_msg_init(const float* __restrict__ bc, float* __restrict__ msg, size_t total) {
    size_t i = (size_t)blockIdx.x * blockDim.x + threadIdx.x;
    if (i < total) msg[i] = bc[i & (GENES - 1)];
}

// wave-per-edge scatter: msg[dst] += dinv[src]*dinv[dst] * x[src]
__global__ void k_edge(const int* __restrict__ ei, const float* __restrict__ dinv,
                       const _Float16* __restrict__ xh, float* __restrict__ msg, int E) {
    int w    = (int)(((size_t)blockIdx.x * blockDim.x + threadIdx.x) >> 5);
    int lane = threadIdx.x & 31;
    if (w >= E) return;
    const int src = ei[w];
    const int dst = ei[E + w];
    const float nrm = dinv[src] * dinv[dst];
    const _Float16* xr = xh + (size_t)src * GENES;
    float* mr = msg + (size_t)dst * GENES;
#pragma unroll
    for (int t = 0; t < 16; ++t) {
        int g = lane + t * 32;
        atomicAdd(&mr[g], nrm * (float)xr[g]);
    }
}

// ---------------- column sums for the mean ----------------
__global__ void k_colsum(const _Float16* __restrict__ gexh, float* __restrict__ colsum,
                         int nspots, int rows_per_block) {
    int g  = threadIdx.x;                  // blockDim.x == GENES
    int r0 = blockIdx.x * rows_per_block;
    int r1 = r0 + rows_per_block; if (r1 > nspots) r1 = nspots;
    float s = 0.0f;
    for (int r = r0; r < r1; ++r) s += (float)gexh[(size_t)r * GENES + g];
    atomicAdd(&colsum[g], s);
}

// ---------------- mean-field log partition function (single block) ----------------
__global__ void k_final(const float* __restrict__ colsum, const float* __restrict__ Wc,
                        const float* __restrict__ Wl, float* __restrict__ outScalar, int nspots) {
    __shared__ float smean[GENES];
    __shared__ float red1[GENES];
    __shared__ float red2[GENES];
    const int g = threadIdx.x;
    const float Nf = (float)nspots;
    smean[g] = colsum[g] / Nf;
    __syncthreads();

    float a1 = 0.0f, a2 = 0.0f;
    for (int k = 0; k < GENES; ++k) {
        float wc = Wc[(size_t)g * GENES + k];
        float wl = Wl[(size_t)g * GENES + k];
        float mk = smean[k];
        a1 += (6.0f * wc + 2.0f * wl) * mk;   // (N_NEIGHBORS*Wc + 2*Wl) @ mean
        a2 += (wl + 3.0f * wc) * mk;          // (Wl + 0.5*N_NEIGHBORS*Wc) @ mean
    }
    red1[g] = a1 * a1;
    red2[g] = smean[g] * a2;
    __syncthreads();
    for (int s = GENES / 2; s > 0; s >>= 1) {
        if (g < s) { red1[g] += red1[g + s]; red2[g] += red2[g + s]; }
        __syncthreads();
    }
    if (g == 0) {
        float gn = sqrtf(red1[0]);
        float z_mean = -Nf * red2[0];
        float gc = fminf(gn, 20.0f);
        float z_int = (gn > 20.0f) ? Nf * (gn - logf(gn))
                                   : Nf * logf((expf(gc) - expf(-gc)) / gc);
        *outScalar = z_mean + z_int;
    }
}

extern "C" void kernel_launch(void* const* d_in, const int* in_sizes, int n_in,
                              void* d_out, int out_size, void* d_ws, size_t ws_size,
                              hipStream_t stream) {
    (void)n_in; (void)out_size; (void)ws_size;
    const float* sphex = (const float*)d_in[0];
    const float* Wc    = (const float*)d_in[1];
    const float* bc    = (const float*)d_in[2];
    const float* Wl    = (const float*)d_in[3];
    const float* bl    = (const float*)d_in[4];
    const int*   ei    = (const int*)d_in[5];

    const int G = GENES;
    const int nspots = in_sizes[0] / (G - 1);
    const int E      = in_sizes[5] / 2;

    float* msg       = (float*)d_out;
    float* msg_intra = msg + (size_t)nspots * G;
    float* outScalar = msg + 2 * (size_t)nspots * G;

    // workspace carve-out (256B aligned)
    char* ws = (char*)d_ws;
    size_t off = 0;
    auto carve = [&](size_t bytes) -> char* {
        char* p = ws + off;
        off = (off + bytes + 255) & ~(size_t)255;
        return p;
    };
    _Float16* gexh = (_Float16*)carve((size_t)nspots * G * sizeof(_Float16));
    _Float16* xh   = (_Float16*)carve((size_t)nspots * G * sizeof(_Float16));
    _Float16* Wch  = (_Float16*)carve((size_t)G * G * sizeof(_Float16));
    _Float16* Wlh  = (_Float16*)carve((size_t)G * G * sizeof(_Float16));
    float*    deg    = (float*)carve((size_t)nspots * sizeof(float));
    float*    dinv   = (float*)carve((size_t)nspots * sizeof(float));
    float*    colsum = (float*)carve((size_t)G * sizeof(float));

    // 1) weights -> f16
    const int wn = G * G;
    k_cvt_f16<<<(wn + 255) / 256, 256, 0, stream>>>(Wc, Wch, wn);
    k_cvt_f16<<<(wn + 255) / 256, 256, 0, stream>>>(Wl, Wlh, wn);

    // 2) gex (f16), one wave per row, 8 waves per block
    k_gex<<<(nspots + 7) / 8, 256, 0, stream>>>(sphex, gexh, nspots);

    // 3) GEMMs via WMMA: x = gex @ Wc^T (f16 scratch); msg_intra = gex @ Wl^T + bl (f32 output)
    dim3 gg((nspots + 15) / 16, G / 256);
    k_gemm_wmma<_Float16><<<gg, 128, 0, stream>>>(gexh, Wch, nullptr, xh, nspots);
    k_gemm_wmma<float><<<gg, 128, 0, stream>>>(gexh, Wlh, bl, msg_intra, nspots);

    // 4) degrees and symmetric norm
    k_zero<<<(nspots + 255) / 256, 256, 0, stream>>>(deg, nspots);
    k_deg<<<(E + 255) / 256, 256, 0, stream>>>(ei + E, deg, E);
    k_dinv<<<(nspots + 255) / 256, 256, 0, stream>>>(deg, dinv, nspots);

    // 5) msg = bc (broadcast), then scatter-add edges
    size_t msgel = (size_t)nspots * G;
    k_msg_init<<<(unsigned)((msgel + 255) / 256), 256, 0, stream>>>(bc, msg, msgel);
    k_edge<<<(E + 7) / 8, 256, 0, stream>>>(ei, dinv, xh, msg, E);

    // 6) mean + log partition scalar
    k_zero<<<(G + 255) / 256, 256, 0, stream>>>(colsum, G);
    const int rpb = 400;
    k_colsum<<<(nspots + rpb - 1) / rpb, G, 0, stream>>>(gexh, colsum, nspots, rpb);
    k_final<<<1, G, 0, stream>>>(colsum, Wc, Wl, outScalar, nspots);
}
